// GATRotationRegressor_19954418057701
// MI455X (gfx1250) — compile-verified, matrix operand order checked by reading, not run on hardware
//
#include <hip/hip_runtime.h>

// ---------------- problem constants ----------------
#define BQ     4              // batch elements per workgroup
#define NJ     24             // joints
#define RWS    (BQ * NJ)      // 96 rows per workgroup
#define HD     128            // hidden dim
#define HS     136            // padded f16 LDS stride (halves)
#define TS     129            // padded f32 LDS stride (floats)
#define NHEADS 4
#define CH     32             // per-head channels
#define NL     3              // GAT layers
#define HHALF  64             // MLP hidden
#define OUTD   6
#define MAXDEG 5
#define BTOT   16384

typedef _Float16 half_t;
typedef __attribute__((ext_vector_type(16))) _Float16 v16h;
typedef __attribute__((ext_vector_type(8)))  _Float16 v8h;
typedef __attribute__((ext_vector_type(8)))  float    v8f;

// In-neighbor lists derived from PARENTS (skeleton edges both dirs + self loops).
__constant__ int IN_DEG[NJ] = {4,3,3,3,3,3,3,3,3,5,2,2,3,3,3,2,3,3,3,3,3,3,2,2};
__constant__ int IN_NBR[NJ][MAXDEG] = {
  {1,2,3,0,0},   {0,4,1,0,0},   {0,5,2,0,0},   {0,6,3,0,0},
  {1,7,4,0,0},   {2,8,5,0,0},   {3,9,6,0,0},   {4,10,7,0,0},
  {5,11,8,0,0},  {6,12,13,14,9},{7,10,0,0,0},  {8,11,0,0,0},
  {9,15,12,0,0}, {9,16,13,0,0}, {9,17,14,0,0}, {12,15,0,0,0},
  {13,18,16,0,0},{14,19,17,0,0},{16,20,18,0,0},{17,21,19,0,0},
  {18,22,20,0,0},{19,23,21,0,0},{20,22,0,0,0}, {21,23,0,0,0}};

// Load one 16x32 f16 WMMA A/B fragment from LDS (row-major, stride HS).
// ISA layout: lanes 0-15 hold rows M=0..15 with K = {0..7,16..23}+kbase,
// lanes 16-31 hold the same rows with K = {8..15,24..31}+kbase.
__device__ __forceinline__ v16h load_frag(const half_t* __restrict__ base,
                                          int lane, int kbase) {
  const int r   = lane & 15;
  const int off = (lane >> 4) ? 8 : 0;
  const half_t* p = base + r * HS + kbase + off;
  v8h lo = *(const v8h*)(p);
  v8h hi = *(const v8h*)(p + 16);
  v16h a;
#pragma unroll
  for (int i = 0; i < 8; ++i) { a[i] = lo[i]; a[i + 8] = hi[i]; }
  return a;
}

__device__ __forceinline__ float wave_sum(float v) {
#pragma unroll
  for (int m = 16; m; m >>= 1) v += __shfl_xor(v, m, 32);
  return v;
}

extern "C" __global__ void __launch_bounds__(256)
gat_rotation_fused(const float* __restrict__ x,
                   const float* __restrict__ in_w,  const float* __restrict__ in_b,
                   const float* __restrict__ res_w, const float* __restrict__ res_b,
                   const float* __restrict__ pos,
                   const float* __restrict__ gat_w,
                   const float* __restrict__ att_s, const float* __restrict__ att_d,
                   const float* __restrict__ gat_b,
                   const float* __restrict__ ln_g,  const float* __restrict__ ln_b,
                   const float* __restrict__ w1,    const float* __restrict__ b1,
                   const float* __restrict__ lng2,  const float* __restrict__ lnb2,
                   const float* __restrict__ w2,    const float* __restrict__ b2,
                   float* __restrict__ out) {
  __shared__ __align__(16) half_t hS [RWS * HS];   // h, f16, padded
  __shared__ __align__(16) half_t xlS[RWS * HS];   // xl = h @ W, f16
  __shared__ __align__(16) half_t wS [HD  * HS];   // transposed weights, f16
  __shared__ __align__(16) float  tmpS[RWS * TS];  // f32 scratch (elu/LN, y1)
  __shared__ float aSs [RWS * NHEADS];
  __shared__ float aDs [RWS * NHEADS];
  __shared__ float alpS[BQ * NJ * NHEADS * MAXDEG];
  __shared__ float xS  [RWS * 3];
  __shared__ float attSl[NHEADS * CH];
  __shared__ float attDl[NHEADS * CH];

  const int tid  = threadIdx.x;
  const int wave = tid >> 5;
  const int lane = tid & 31;
  const long long rowBase = (long long)blockIdx.x * RWS;

  // ---- stage x tile ----
  for (int i = tid; i < RWS * 3; i += 256) xS[i] = x[rowBase * 3 + i];
  __syncthreads();

  // ---- h = x @ in_w + in_b + pos ----
  for (int i = tid; i < RWS * HD; i += 256) {
    const int r = i >> 7, c = i & 127;
    const int j = r % NJ;
    float v = xS[r*3+0] * in_w[c] + xS[r*3+1] * in_w[HD + c] +
              xS[r*3+2] * in_w[2*HD + c] + in_b[c] + pos[j*HD + c];
    hS[r * HS + c] = (half_t)v;
  }
  __syncthreads();

  // =================== GAT layers ===================
  for (int l = 0; l < NL; ++l) {
    // ---- stage W^T (f16) + attention vectors ----
    const float* W  = gat_w + l * HD * HD;
    const float* As = att_s + l * NHEADS * CH;
    const float* Ad = att_d + l * NHEADS * CH;
    for (int i = tid; i < HD * HD; i += 256) {
      const int k = i >> 7, n = i & 127;
      wS[n * HS + k] = (half_t)W[k * HD + n];
    }
    if (tid < NHEADS * CH) { attSl[tid] = As[tid]; attDl[tid] = Ad[tid]; }
    __syncthreads();

    // prefetch next layer's weights into cache while WMMA runs
    if (l + 1 < NL) __builtin_prefetch(W + HD * HD + tid * 64, 0, 1);

    // ---- xl = h @ W via WMMA (per-wave: fixed ntile, sweep 6 mtiles) ----
    {
      const int nt = wave;                  // 8 waves == 8 N-tiles
      v16h b0 = load_frag(wS + nt * 16 * HS, lane, 0);
      v16h b1f = load_frag(wS + nt * 16 * HS, lane, 32);
      v16h b2f = load_frag(wS + nt * 16 * HS, lane, 64);
      v16h b3f = load_frag(wS + nt * 16 * HS, lane, 96);
      for (int mt = 0; mt < RWS / 16; ++mt) {
        v8f acc = {};
        acc = __builtin_amdgcn_wmma_f32_16x16x32_f16(
            false, load_frag(hS + mt * 16 * HS, lane, 0),  false, b0,  (short)0, acc, false, false);
        acc = __builtin_amdgcn_wmma_f32_16x16x32_f16(
            false, load_frag(hS + mt * 16 * HS, lane, 32), false, b1f, (short)0, acc, false, false);
        acc = __builtin_amdgcn_wmma_f32_16x16x32_f16(
            false, load_frag(hS + mt * 16 * HS, lane, 64), false, b2f, (short)0, acc, false, false);
        acc = __builtin_amdgcn_wmma_f32_16x16x32_f16(
            false, load_frag(hS + mt * 16 * HS, lane, 96), false, b3f, (short)0, acc, false, false);
        const int col   = nt * 16 + (lane & 15);
        const int rbase = mt * 16 + ((lane >> 4) * 8);
#pragma unroll
        for (int r = 0; r < 8; ++r)
          xlS[(rbase + r) * HS + col] = (half_t)acc[r];
      }
    }
    __syncthreads();

    // ---- attention logits a_s, a_d ----
    for (int t = tid; t < RWS * NHEADS; t += 256) {
      const int r = t >> 2, hd = t & 3;
      const half_t* xr = xlS + r * HS + hd * CH;
      float s0 = 0.f, s1 = 0.f;
#pragma unroll
      for (int c = 0; c < CH; ++c) {
        const float v = (float)xr[c];
        s0 += v * attSl[hd * CH + c];
        s1 += v * attDl[hd * CH + c];
      }
      aSs[t] = s0; aDs[t] = s1;
    }
    __syncthreads();

    // ---- per-destination edge softmax (<=5 in-neighbors) ----
    for (int t = tid; t < BQ * NJ * NHEADS; t += 256) {
      const int hd = t & 3;
      const int j  = (t >> 2) % NJ;
      const int b  = t / (NJ * NHEADS);
      const int d  = IN_DEG[j];
      const float ad = aDs[(b * NJ + j) * NHEADS + hd];
      float e[MAXDEG];
      float m = -1e30f;
      for (int s = 0; s < d; ++s) {
        float v = aSs[(b * NJ + IN_NBR[j][s]) * NHEADS + hd] + ad;
        v = v > 0.f ? v : 0.2f * v;           // leaky_relu(0.2)
        e[s] = v; m = fmaxf(m, v);
      }
      float den = 0.f;
      for (int s = 0; s < d; ++s) { e[s] = __expf(e[s] - m); den += e[s]; }
      const float inv = 1.f / den;
      for (int s = 0; s < d; ++s) alpS[t * MAXDEG + s] = e[s] * inv;
    }
    __syncthreads();

    // ---- aggregate messages + bias + ELU -> tmpS (f32) ----
    const float* Gb = gat_b + l * HD;
    for (int i = tid; i < RWS * HD; i += 256) {
      const int r = i >> 7, c = i & 127;
      const int b = r / NJ, j = r % NJ, hd = c >> 5;
      const int d = IN_DEG[j];
      const float* al = alpS + ((b * NJ + j) * NHEADS + hd) * MAXDEG;
      float acc = Gb[c];
      for (int s = 0; s < d; ++s)
        acc += al[s] * (float)xlS[(b * NJ + IN_NBR[j][s]) * HS + c];
      acc = acc > 0.f ? acc : (__expf(acc) - 1.f);   // elu
      tmpS[r * TS + c] = acc;
    }
    __syncthreads();

    // ---- LayerNorm + residual -> h (wave-parallel, 4 channels/lane) ----
    const float* G  = ln_g + l * HD;
    const float* Bt = ln_b + l * HD;
    for (int r = wave; r < RWS; r += 8) {
      const float v0 = tmpS[r * TS + lane];
      const float v1 = tmpS[r * TS + lane + 32];
      const float v2 = tmpS[r * TS + lane + 64];
      const float v3 = tmpS[r * TS + lane + 96];
      const float mean = wave_sum(v0 + v1 + v2 + v3) * (1.f / HD);
      const float d0 = v0 - mean, d1 = v1 - mean, d2 = v2 - mean, d3 = v3 - mean;
      const float var = wave_sum(d0*d0 + d1*d1 + d2*d2 + d3*d3) * (1.f / HD);
      const float rs = rsqrtf(var + 1e-5f);
      half_t* hr = hS + r * HS;
#pragma unroll
      for (int q = 0; q < 4; ++q) {
        const int c = lane + q * 32;
        const float dv = (q == 0 ? d0 : q == 1 ? d1 : q == 2 ? d2 : d3);
        float y = dv * rs * G[c] + Bt[c];
        if (l > 0) y += (float)hr[c];
        hr[c] = (half_t)y;
      }
    }
    __syncthreads();
  }

  // ---- h += res (recomputed from x) ----
  for (int i = tid; i < RWS * HD; i += 256) {
    const int r = i >> 7, c = i & 127;
    float res = xS[r*3+0] * res_w[c] + xS[r*3+1] * res_w[HD + c] +
                xS[r*3+2] * res_w[2*HD + c] + res_b[c];
    hS[r * HS + c] = (half_t)((float)hS[r * HS + c] + res);
  }
  __syncthreads();

  // ---- stage w1^T ----
  for (int i = tid; i < HD * HHALF; i += 256) {
    const int k = i >> 6, n = i & 63;
    wS[n * HS + k] = (half_t)w1[k * HHALF + n];
  }
  __syncthreads();

  // ---- y1 = relu(h @ w1 + b1) via WMMA (per-wave: fixed ntile) ----
  {
    const int nt = wave & 3;                 // 4 N-tiles, waves pair up on mtiles
    v16h b0 = load_frag(wS + nt * 16 * HS, lane, 0);
    v16h b1f = load_frag(wS + nt * 16 * HS, lane, 32);
    v16h b2f = load_frag(wS + nt * 16 * HS, lane, 64);
    v16h b3f = load_frag(wS + nt * 16 * HS, lane, 96);
    for (int mt = (wave >> 2); mt < RWS / 16; mt += 2) {
      v8f acc = {};
      acc = __builtin_amdgcn_wmma_f32_16x16x32_f16(
          false, load_frag(hS + mt * 16 * HS, lane, 0),  false, b0,  (short)0, acc, false, false);
      acc = __builtin_amdgcn_wmma_f32_16x16x32_f16(
          false, load_frag(hS + mt * 16 * HS, lane, 32), false, b1f, (short)0, acc, false, false);
      acc = __builtin_amdgcn_wmma_f32_16x16x32_f16(
          false, load_frag(hS + mt * 16 * HS, lane, 64), false, b2f, (short)0, acc, false, false);
      acc = __builtin_amdgcn_wmma_f32_16x16x32_f16(
          false, load_frag(hS + mt * 16 * HS, lane, 96), false, b3f, (short)0, acc, false, false);
      const int col   = nt * 16 + (lane & 15);
      const int rbase = mt * 16 + ((lane >> 4) * 8);
#pragma unroll
      for (int r = 0; r < 8; ++r) {
        float v = acc[r] + b1[col];
        tmpS[(rbase + r) * TS + col] = v > 0.f ? v : 0.f;
      }
    }
  }
  __syncthreads();

  // ---- LayerNorm over 64 (wave-parallel, 2 channels/lane) ----
  for (int r = wave; r < RWS; r += 8) {
    const float v0 = tmpS[r * TS + lane];
    const float v1 = tmpS[r * TS + lane + 32];
    const float mean = wave_sum(v0 + v1) * (1.f / HHALF);
    const float d0 = v0 - mean, d1 = v1 - mean;
    const float var = wave_sum(d0*d0 + d1*d1) * (1.f / HHALF);
    const float rs = rsqrtf(var + 1e-5f);
    tmpS[r * TS + lane]      = d0 * rs * lng2[lane]      + lnb2[lane];
    tmpS[r * TS + lane + 32] = d1 * rs * lng2[lane + 32] + lnb2[lane + 32];
  }
  __syncthreads();

  // ---- final linear [64 -> 6] ----
  for (int t = tid; t < RWS * OUTD; t += 256) {
    const int r = t / OUTD, o = t % OUTD;
    float acc = b2[o];
#pragma unroll 8
    for (int k = 0; k < HHALF; ++k) acc += tmpS[r * TS + k] * w2[k * OUTD + o];
    out[(rowBase + r) * OUTD + o] = acc;
  }
}

extern "C" void kernel_launch(void* const* d_in, const int* in_sizes, int n_in,
                              void* d_out, int out_size, void* d_ws, size_t ws_size,
                              hipStream_t stream) {
  (void)in_sizes; (void)n_in; (void)out_size; (void)d_ws; (void)ws_size;
  const float* x     = (const float*)d_in[0];
  const float* in_w  = (const float*)d_in[1];
  const float* in_b  = (const float*)d_in[2];
  const float* res_w = (const float*)d_in[3];
  const float* res_b = (const float*)d_in[4];
  const float* pos   = (const float*)d_in[5];
  const float* gat_w = (const float*)d_in[6];
  const float* att_s = (const float*)d_in[7];
  const float* att_d = (const float*)d_in[8];
  const float* gat_b = (const float*)d_in[9];
  const float* ln_g  = (const float*)d_in[10];
  const float* ln_b  = (const float*)d_in[11];
  const float* w1    = (const float*)d_in[12];
  const float* b1    = (const float*)d_in[13];
  const float* lng2  = (const float*)d_in[14];
  const float* lnb2  = (const float*)d_in[15];
  const float* w2    = (const float*)d_in[16];
  const float* b2    = (const float*)d_in[17];
  float* out = (float*)d_out;

  dim3 grid(BTOT / BQ), block(256);
  gat_rotation_fused<<<grid, block, 0, stream>>>(
      x, in_w, in_b, res_w, res_b, pos, gat_w, att_s, att_d, gat_b,
      ln_g, ln_b, w1, b1, lng2, lnb2, w2, b2, out);
}
